// DirectG2PNEFS_62886911148203
// MI455X (gfx1250) — compile-verified, matrix operand order checked by reading, not run on hardware
//
#include <hip/hip_runtime.h>
#include <hip/hip_bf16.h>
#include <hip/hip_fp16.h>

typedef __attribute__((ext_vector_type(16))) _Float16 v16h;
typedef __attribute__((ext_vector_type(8)))  _Float16 v8h;
typedef __attribute__((ext_vector_type(8)))  float    v8f;

#define BB     32
#define HDIM   256
#define TLEN   256
#define NEFS   128
#define GATES  1024
#define VOCAB  256

// ---------------------------------------------------------------------------
// WMMA helpers (CDNA5 gfx1250, wave32)
// ---------------------------------------------------------------------------
__device__ __forceinline__ v8f wmma_f16(v16h a, v16h b, v8f c) {
  // D = A(16x32 f16) * B(32x16 f16) + C(16x16 f32)
  return __builtin_amdgcn_wmma_f32_16x16x32_f16(
      /*neg_a=*/false, a, /*neg_b=*/false, b,
      /*c_mod=*/(short)0, c, /*reuse_a=*/false, /*reuse_b=*/false);
}

// Load one 16x32 f16 fragment (works for both A and B operands).
// src layout: row-major [rows, ld] f16.  For A: rows = M; for B (weights
// stored as (N,K) row-major) rows = N, and the fragment is logically K x N.
// ISA layout: lanes 0-15 hold K = k0+0..7 and k0+16..23; lanes 16-31 hold
// K = k0+8..15 and k0+24..31; row = row0 + (lane & 15).
__device__ __forceinline__ v16h load_frag16(const _Float16* base, int row0,
                                            int ld, int k0) {
  int lane = threadIdx.x & 31;
  int half = lane >> 4;
  int r = row0 + (lane & 15);
  const _Float16* p = base + (size_t)r * ld + k0 + half * 8;
  union { v16h v; v8h h[2]; } u;
  u.h[0] = *(const v8h*)(p);
  u.h[1] = *(const v8h*)(p + 16);
  return u.v;
}

__device__ __forceinline__ float sigf(float x) {
  return 1.0f / (1.0f + expf(-x));
}

// ---------------------------------------------------------------------------
// Small utility kernels
// ---------------------------------------------------------------------------
__global__ void f32_to_f16_kernel(const float* __restrict__ s,
                                  _Float16* __restrict__ d, int n) {
  int i = blockIdx.x * blockDim.x + threadIdx.x;
  if (i < n) d[i] = (_Float16)s[i];
}

__global__ void add_bias_kernel(const float* __restrict__ a,
                                const float* __restrict__ b,
                                float* __restrict__ o, int n) {
  int i = blockIdx.x * blockDim.x + threadIdx.x;
  if (i < n) o[i] = a[i] + b[i];
}

__global__ void embed_text_kernel(const int* __restrict__ text,
                                  const _Float16* __restrict__ emb16,
                                  _Float16* __restrict__ x0) {
  int i = blockIdx.x * blockDim.x + threadIdx.x;  // over 8192*256
  if (i < BB * TLEN * HDIM) {
    int r = i >> 8;          // b*T + t
    int k = i & 255;
    x0[i] = emb16[text[r] * HDIM + k];
  }
}

// ---------------------------------------------------------------------------
// Input projection: XP[r,n] = sum_k X[r,k] * W[n,k] + bias[n]
// X: (8192, K) f16 row-major. W: (1024, K) f16 row-major. XP: (8192,1024) f32
// grid: (8192/16, 2), block 256 (8 waves). Each wave: 1 M-tile x 4 N-tiles.
// ---------------------------------------------------------------------------
__global__ __launch_bounds__(256)
void gemm_inproj_kernel(const _Float16* __restrict__ X,
                        const _Float16* __restrict__ W,
                        const float* __restrict__ bias,
                        float* __restrict__ XP, int K) {
  int wave = threadIdx.x >> 5;
  int lane = threadIdx.x & 31;
  int nl = lane & 15, half = lane >> 4;
  int m0 = blockIdx.x * 16;
  int n0 = blockIdx.y * 512 + wave * 64;

  v8f acc[4];
  for (int s = 0; s < 4; ++s) {
    float bv = bias[n0 + s * 16 + nl];
#pragma unroll
    for (int j = 0; j < 8; ++j) acc[s][j] = bv;
  }

  int nk = K >> 5;
  for (int kc = 0; kc < nk; ++kc) {
    v16h a = load_frag16(X, m0, K, kc * 32);
#pragma unroll
    for (int s = 0; s < 4; ++s) {
      v16h b = load_frag16(W, n0 + s * 16, K, kc * 32);
      acc[s] = wmma_f16(a, b, acc[s]);
    }
  }

  int mrow = m0 + 8 * half;
  for (int s = 0; s < 4; ++s)
#pragma unroll
    for (int j = 0; j < 8; ++j)
      XP[(size_t)(mrow + j) * GATES + n0 + s * 16 + nl] = acc[s][j];
}

// ---------------------------------------------------------------------------
// Persistent recurrent LSTM over T steps. One block, 1024 threads = 32 waves.
// Wave w: m_tile = w&1 (batch rows), p = w>>1 selects hidden cols p*16..+15.
// Its 4 accumulators are the i/f/g/o gate blocks for the SAME (m,n) set, so
// the LSTM update is lane-local; c stays in registers for all T steps.
// h (32x256 f16) lives in LDS; W_hh streams from L2 each step.
// ---------------------------------------------------------------------------
__global__ __launch_bounds__(1024)
void lstm_rec_kernel(const float* __restrict__ XP,        // (B*T,1024) +bias
                     const _Float16* __restrict__ Whh,    // (1024,256)
                     _Float16* __restrict__ Y,            // (B*T, ldy)
                     int ldy, int coloff, int reverse) {
  __shared__ _Float16 hbuf[BB * HDIM];
  int tid = threadIdx.x, wave = tid >> 5, lane = tid & 31;
  int mt = wave & 1, p = wave >> 1;
  int nl = lane & 15, half = lane >> 4;
  int mbase = mt * 16 + 8 * half;   // + j = batch row
  int n_h = p * 16 + nl;            // hidden index

  for (int i = tid; i < BB * HDIM; i += 1024) hbuf[i] = (_Float16)0.0f;
  float c[8];
#pragma unroll
  for (int j = 0; j < 8; ++j) c[j] = 0.0f;
  __syncthreads();

  for (int s = 0; s < TLEN; ++s) {
    int t = reverse ? (TLEN - 1 - s) : s;
    v8f acc[4];
#pragma unroll
    for (int b = 0; b < 4; ++b)
#pragma unroll
      for (int j = 0; j < 8; ++j)
        acc[b][j] = XP[(size_t)((mbase + j) * TLEN + t) * GATES + b * 256 + n_h];

#pragma unroll
    for (int kc = 0; kc < 8; ++kc) {
      v16h a = load_frag16(hbuf, mt * 16, HDIM, kc * 32);
#pragma unroll
      for (int b = 0; b < 4; ++b) {
        v16h w = load_frag16(Whh, b * 256 + p * 16, HDIM, kc * 32);
        acc[b] = wmma_f16(a, w, acc[b]);
      }
    }
    __syncthreads();  // all waves done reading old h

#pragma unroll
    for (int j = 0; j < 8; ++j) {
      float iv = sigf(acc[0][j]);
      float fv = sigf(acc[1][j]);
      float gv = tanhf(acc[2][j]);
      float ov = sigf(acc[3][j]);
      c[j] = fv * c[j] + iv * gv;
      float h = ov * tanhf(c[j]);
      hbuf[(mbase + j) * HDIM + n_h] = (_Float16)h;
      Y[(size_t)((mbase + j) * TLEN + t) * ldy + coloff + n_h] = (_Float16)h;
    }
    __syncthreads();  // new h visible for next step
  }
}

// ---------------------------------------------------------------------------
// One decoder LSTM layer (device helper so no pointer tables of __shared__
// addresses are materialized as static initializers).
// Reads input activations `in` (B x K f16) and recurrent state `hb`
// (B x 256 f16), updates per-lane cell state c[8] and writes new h to hb.
// ---------------------------------------------------------------------------
__device__ __forceinline__ void dec_lstm_layer(
    const _Float16* in, int K,
    const _Float16* __restrict__ wih, const _Float16* __restrict__ whh,
    const float* __restrict__ bias,
    _Float16* hb, float* c,
    int mt, int p, int nl, int mbase, int n_h) {
  v8f acc[4];
  for (int s = 0; s < 4; ++s) {
    float bv = bias[s * 256 + p * 16 + nl];
#pragma unroll
    for (int j = 0; j < 8; ++j) acc[s][j] = bv;
  }
  int nk = K >> 5;
  for (int kc = 0; kc < nk; ++kc) {            // input contribution
    v16h a = load_frag16(in, mt * 16, K, kc * 32);
#pragma unroll
    for (int s = 0; s < 4; ++s) {
      v16h b = load_frag16(wih, s * 256 + p * 16, K, kc * 32);
      acc[s] = wmma_f16(a, b, acc[s]);
    }
  }
#pragma unroll
  for (int kc = 0; kc < 8; ++kc) {             // recurrent contribution
    v16h a = load_frag16(hb, mt * 16, HDIM, kc * 32);
#pragma unroll
    for (int s = 0; s < 4; ++s) {
      v16h b = load_frag16(whh, s * 256 + p * 16, HDIM, kc * 32);
      acc[s] = wmma_f16(a, b, acc[s]);
    }
  }
  __syncthreads();  // all waves finished reading h (and `in`)
#pragma unroll
  for (int j = 0; j < 8; ++j) {
    float iv = sigf(acc[0][j]);
    float fv = sigf(acc[1][j]);
    float gv = tanhf(acc[2][j]);
    float ov = sigf(acc[3][j]);
    c[j] = fv * c[j] + iv * gv;
    float h = ov * tanhf(c[j]);
    hb[(mbase + j) * HDIM + n_h] = (_Float16)h;
  }
  __syncthreads();
}

// ---------------------------------------------------------------------------
// Persistent greedy decoder: 128 steps x (3 LSTM layers + WMMA projection +
// argmax + embedding re-lookup). One block, 1024 threads.
// ---------------------------------------------------------------------------
__global__ __launch_bounds__(1024)
void decoder_kernel(const _Float16* __restrict__ encY,   // (B*T, 512) f16
                    const _Float16* __restrict__ wih0, const _Float16* __restrict__ whh0, const float* __restrict__ bb0,
                    const _Float16* __restrict__ wih1, const _Float16* __restrict__ whh1, const float* __restrict__ bb1,
                    const _Float16* __restrict__ wih2, const _Float16* __restrict__ whh2, const float* __restrict__ bb2,
                    const _Float16* __restrict__ projw,  // (256,256) f16
                    const float* __restrict__ projb,     // (256) f32
                    const _Float16* __restrict__ emb16,  // (256,256) f16
                    float* __restrict__ out) {           // (B, NEFS, 256) f32
  __shared__ _Float16 inp0[BB * 512];
  __shared__ _Float16 hb0[BB * HDIM];
  __shared__ _Float16 hb1[BB * HDIM];
  __shared__ _Float16 hb2[BB * HDIM];
  __shared__ float logits[BB * VOCAB];
  __shared__ int tok[BB];

  int tid = threadIdx.x, wave = tid >> 5, lane = tid & 31;
  int mt = wave & 1, p = wave >> 1;
  int nl = lane & 15, half = lane >> 4;
  int mbase = mt * 16 + 8 * half;
  int n_h = p * 16 + nl;

  // init: inp0 = encoder output at t=0 (B x 512); h = 0; c = 0
  for (int i = tid; i < BB * 512; i += 1024)
    inp0[i] = encY[(size_t)((i >> 9) * TLEN) * 512 + (i & 511)];
  for (int i = tid; i < BB * HDIM; i += 1024) {
    hb0[i] = (_Float16)0.0f; hb1[i] = (_Float16)0.0f; hb2[i] = (_Float16)0.0f;
  }
  float c0[8], c1[8], c2[8];
#pragma unroll
  for (int j = 0; j < 8; ++j) { c0[j] = 0.0f; c1[j] = 0.0f; c2[j] = 0.0f; }
  __syncthreads();

  for (int t = 0; t < NEFS; ++t) {
    // ---- 3 LSTM layers ----
    dec_lstm_layer(inp0, 512, wih0, whh0, bb0, hb0, c0, mt, p, nl, mbase, n_h);
    dec_lstm_layer(hb0,  256, wih1, whh1, bb1, hb1, c1, mt, p, nl, mbase, n_h);
    dec_lstm_layer(hb1,  256, wih2, whh2, bb2, hb2, c2, mt, p, nl, mbase, n_h);

    // ---- projection: logits = h2 @ projw^T + projb; 32 tiles / 32 waves ----
    {
      int nt = wave >> 1;               // 0..15, n0 = nt*16
      v8f acc;
      float bv = projb[nt * 16 + nl];
#pragma unroll
      for (int j = 0; j < 8; ++j) acc[j] = bv;
#pragma unroll
      for (int kc = 0; kc < 8; ++kc) {
        v16h a = load_frag16(hb2, mt * 16, HDIM, kc * 32);
        v16h b = load_frag16(projw, nt * 16, HDIM, kc * 32);
        acc = wmma_f16(a, b, acc);
      }
#pragma unroll
      for (int j = 0; j < 8; ++j) {
        int m = mbase + j;
        int n = nt * 16 + nl;
        logits[m * VOCAB + n] = acc[j];
        out[((size_t)m * NEFS + t) * VOCAB + n] = acc[j];
      }
    }
    __syncthreads();

    // ---- argmax (first max, like jnp.argmax) ----
    if (tid < BB) {
      float best = logits[tid * VOCAB];
      int bi = 0;
      for (int n = 1; n < VOCAB; ++n) {
        float v = logits[tid * VOCAB + n];
        if (v > best) { best = v; bi = n; }
      }
      tok[tid] = bi;
    }
    __syncthreads();

    // ---- next input = concat(emb, emb) ----
    for (int i = tid; i < BB * HDIM; i += 1024) {
      int m = i >> 8, n = i & 255;
      _Float16 e = emb16[tok[m] * HDIM + n];
      inp0[m * 512 + n] = e;
      inp0[m * 512 + 256 + n] = e;
    }
    __syncthreads();
  }
}

// ---------------------------------------------------------------------------
// Host orchestration
// ---------------------------------------------------------------------------
extern "C" void kernel_launch(void* const* d_in, const int* in_sizes, int n_in,
                              void* d_out, int out_size, void* d_ws, size_t ws_size,
                              hipStream_t stream) {
  (void)in_sizes; (void)n_in; (void)out_size; (void)ws_size;

  const int*   text      = (const int*)d_in[0];
  // d_in[1] = max_nefs_len (always 128, hardcoded as NEFS)
  const float* embedding = (const float*)d_in[2];
  auto encp = [&](int l, int d, int k) { return (const float*)d_in[3 + (l * 2 + d) * 4 + k]; };
  auto decp = [&](int l, int k)        { return (const float*)d_in[27 + l * 4 + k]; };
  const float* proj_w = (const float*)d_in[39];
  const float* proj_b = (const float*)d_in[40];

  char* ws = (char*)d_ws;
  size_t cur = 0;
  auto alloc = [&](size_t bytes) -> char* {
    char* ptr = ws + cur;
    cur = (cur + bytes + 255) & ~(size_t)255;
    return ptr;
  };

  const int encDin[3] = {256, 512, 512};
  const int decDin[3] = {512, 256, 256};

  _Float16* emb16 = (_Float16*)alloc((size_t)VOCAB * HDIM * 2);
  _Float16* x0    = (_Float16*)alloc((size_t)BB * TLEN * HDIM * 2);
  _Float16* ybuf[3];
  for (int l = 0; l < 3; ++l) ybuf[l] = (_Float16*)alloc((size_t)BB * TLEN * 512 * 2);
  float* xp = (float*)alloc((size_t)BB * TLEN * GATES * 4);

  _Float16* ewih[3][2]; _Float16* ewhh[3][2]; float* ebias[3][2];
  for (int l = 0; l < 3; ++l)
    for (int d = 0; d < 2; ++d) {
      ewih[l][d]  = (_Float16*)alloc((size_t)GATES * encDin[l] * 2);
      ewhh[l][d]  = (_Float16*)alloc((size_t)GATES * HDIM * 2);
      ebias[l][d] = (float*)alloc((size_t)GATES * 4);
    }
  _Float16* dwih[3]; _Float16* dwhh[3]; float* dbias[3];
  for (int l = 0; l < 3; ++l) {
    dwih[l]  = (_Float16*)alloc((size_t)GATES * decDin[l] * 2);
    dwhh[l]  = (_Float16*)alloc((size_t)GATES * HDIM * 2);
    dbias[l] = (float*)alloc((size_t)GATES * 4);
  }
  _Float16* projw16 = (_Float16*)alloc((size_t)VOCAB * HDIM * 2);

  auto conv = [&](const float* s, _Float16* d, int n) {
    f32_to_f16_kernel<<<(n + 255) / 256, 256, 0, stream>>>(s, d, n);
  };

  // weight conversion + bias combine
  conv(embedding, emb16, VOCAB * HDIM);
  for (int l = 0; l < 3; ++l)
    for (int d = 0; d < 2; ++d) {
      conv(encp(l, d, 0), ewih[l][d], GATES * encDin[l]);
      conv(encp(l, d, 1), ewhh[l][d], GATES * HDIM);
      add_bias_kernel<<<4, 256, 0, stream>>>(encp(l, d, 2), encp(l, d, 3),
                                             ebias[l][d], GATES);
    }
  for (int l = 0; l < 3; ++l) {
    conv(decp(l, 0), dwih[l], GATES * decDin[l]);
    conv(decp(l, 1), dwhh[l], GATES * HDIM);
    add_bias_kernel<<<4, 256, 0, stream>>>(decp(l, 2), decp(l, 3), dbias[l], GATES);
  }
  conv(proj_w, projw16, VOCAB * HDIM);

  // embedding lookup -> x0 (f16)
  embed_text_kernel<<<(BB * TLEN * HDIM + 255) / 256, 256, 0, stream>>>(text, emb16, x0);

  // encoder: 3 bidirectional layers
  for (int l = 0; l < 3; ++l) {
    const _Float16* X = (l == 0) ? x0 : ybuf[l - 1];
    int K = encDin[l];
    for (int d = 0; d < 2; ++d) {
      gemm_inproj_kernel<<<dim3(BB * TLEN / 16, 2), 256, 0, stream>>>(
          X, ewih[l][d], ebias[l][d], xp, K);
      lstm_rec_kernel<<<1, 1024, 0, stream>>>(xp, ewhh[l][d], ybuf[l],
                                              /*ldy=*/512, /*coloff=*/d * 256,
                                              /*reverse=*/d);
    }
  }

  // decoder (persistent, 128 greedy steps)
  decoder_kernel<<<1, 1024, 0, stream>>>(
      ybuf[2],
      dwih[0], dwhh[0], dbias[0],
      dwih[1], dwhh[1], dbias[1],
      dwih[2], dwhh[2], dbias[2],
      projw16, proj_b, emb16, (float*)d_out);
}
